// BoxFuseModule_4458176053253
// MI455X (gfx1250) — matrix operand order repaired and checked by reading, not compiled, not run on hardware
//
#include <hip/hip_runtime.h>

typedef __attribute__((ext_vector_type(16))) __bf16 v16bf;
typedef __attribute__((ext_vector_type(8)))  float  v8f;
typedef __attribute__((ext_vector_type(4)))  float  f32x4;

#define B_    32
#define N_    576
#define L_    100
#define LP_   128     // padded L for WMMA K-dim
#define DLO_  1024
#define DHI_  1536

#if defined(__has_builtin)
#if __has_builtin(__builtin_amdgcn_sched_barrier)
#define SCHED_FENCE() __builtin_amdgcn_sched_barrier(0)
#endif
#endif
#ifndef SCHED_FENCE
#define SCHED_FENCE()
#endif

// ---- helpers -------------------------------------------------------------

__device__ inline unsigned short f2bf(float f) {
    union { float f; unsigned u; } u; u.f = f;
    unsigned r = u.u + 0x7fffu + ((u.u >> 16) & 1u);   // round-to-nearest-even
    return (unsigned short)(r >> 16);
}

__device__ inline float wave_sum(float x) {
    for (int off = 16; off > 0; off >>= 1) x += __shfl_xor(x, off, 32);
    return x;
}

// A-fragment (16x32 bf16, M x K): lane m = l%16, h = l/16.
// Element e holds K = kk + 16h + e.
__device__ inline v16bf load_afrag(const __bf16* tile, int ldk, int kk, int lane) {
    int m = lane & 15, h = lane >> 4;
    const __bf16* p = tile + (size_t)m * ldk + kk + 16 * h;
    union { v16bf v; f32x4 q[2]; } u;
    u.q[0] = *(const f32x4*)(p);
    u.q[1] = *(const f32x4*)(p + 8);
    return u.v;
}

// B-fragment (32x16 bf16, K x N) from K-contiguous ("transposed") storage
// wt[n][K]: lane n = l%16, h = l/16; element e holds K = kk + 16h + e.
__device__ inline v16bf load_bfrag(const __bf16* wt, int ldk, int col0, int kk, int lane) {
    int n = lane & 15, h = lane >> 4;
    return *(const v16bf*)(wt + (size_t)(col0 + n) * ldk + kk + 16 * h);
}

__device__ inline v8f wmma_bf16(v16bf a, v16bf b, v8f c) {
    return __builtin_amdgcn_wmma_f32_16x16x32_bf16(false, a, false, b, (short)0, c, false, false);
}

// ---- kernel: weight transpose + f32->bf16 --------------------------------
__global__ __launch_bounds__(256) void wprep_kernel(const float* __restrict__ w,
                                                    unsigned short* __restrict__ wt,
                                                    int Kd) {
    size_t i = (size_t)blockIdx.x * 256 + threadIdx.x;
    size_t total = (size_t)Kd * 1024;
    if (i >= total) return;
    int k = (int)(i % Kd);
    int n = (int)(i / Kd);
    wt[i] = f2bf(w[(size_t)k * 1024 + n]);
}

// ---- kernel: q projection  (LN(vit) @ Wq + b) -> qbuf bf16 ---------------
// grid: (B*N)/64 blocks, 256 threads (8 waves), dynamic LDS = 64*1024*2 B.
// Wave w owns cols [128w,128w+128); 4 row-subtiles share each B fragment.
__global__ __launch_bounds__(256) void q_proj_kernel(
    const float* __restrict__ x,            // [B*N, 1024]
    const float* __restrict__ lnw, const float* __restrict__ lnb,
    const unsigned short* __restrict__ wt,  // bf16 [1024][1024] K-contig
    const float* __restrict__ bias,
    unsigned short* __restrict__ out)       // bf16 [B*N, 1024]
{
    extern __shared__ char smem[];
    unsigned short* xn = (unsigned short*)smem;   // 64 rows x 1024 bf16 (128 KB)
    const int tid = threadIdx.x, w = tid >> 5, lane = tid & 31;
    const size_t r0 = (size_t)blockIdx.x * 64;

    // LayerNorm: each wave handles 8 rows
    for (int i = 0; i < 8; ++i) {
        int m = 8 * w + i;
        const float* row = x + (r0 + m) * DLO_;
        float s = 0.f, sq = 0.f;
        for (int c = lane; c < DLO_; c += 32) { float t = row[c]; s += t; sq += t * t; }
        s = wave_sum(s); sq = wave_sum(sq);
        float mean = s * (1.f / DLO_);
        float var  = sq * (1.f / DLO_) - mean * mean;
        float rstd = rsqrtf(var + 1e-5f);
        for (int c = lane; c < DLO_; c += 32)
            xn[m * DLO_ + c] = f2bf((row[c] - mean) * rstd * lnw[c] + lnb[c]);
    }
    __syncthreads();

    const __bf16* xb = (const __bf16*)xn;
    const int n = lane & 15, h = lane >> 4;
    for (int ct = 0; ct < 8; ++ct) {
        int col0 = w * 128 + ct * 16;
        v8f a0 = {}, a1 = {}, a2 = {}, a3 = {};
        for (int kk = 0; kk < DLO_; kk += 32) {
            // all fragment loads issue before the scheduling fence; the fence
            // keeps them grouped (distinct regs, one ds wait, WMMAs b2b).
            v16bf b  = load_bfrag((const __bf16*)wt, DLO_, col0, kk, lane);
            v16bf f0 = load_afrag(xb,             DLO_, kk, lane);
            v16bf f1 = load_afrag(xb + 16 * DLO_, DLO_, kk, lane);
            v16bf f2 = load_afrag(xb + 32 * DLO_, DLO_, kk, lane);
            v16bf f3 = load_afrag(xb + 48 * DLO_, DLO_, kk, lane);
            SCHED_FENCE();
            a0 = wmma_bf16(f0, b, a0);
            a1 = wmma_bf16(f1, b, a1);
            a2 = wmma_bf16(f2, b, a2);
            a3 = wmma_bf16(f3, b, a3);
        }
        float bv = bias[col0 + n];
        v8f accs[4] = {a0, a1, a2, a3};
        for (int rt = 0; rt < 4; ++rt)
            for (int e = 0; e < 8; ++e) {
                size_t r = r0 + rt * 16 + e + 8 * h;
                out[r * DLO_ + col0 + n] = f2bf(accs[rt][e] + bv);
            }
    }
}

// ---- kernel: k & v projections -------------------------------------------
// grid: B*4 blocks (4 row-tiles of 32 covering padded rows 0..127 per batch;
// rows >= 100 use zeroed LN input -> padded outputs are bias -> finite).
// dynamic LDS = 32*1536*2 B. Each B fragment feeds 2 row-subtiles.
__global__ __launch_bounds__(256) void kv_proj_kernel(
    const float* __restrict__ box,          // [B, 100, 1536]
    const float* __restrict__ klnw, const float* __restrict__ klnb,
    const unsigned short* __restrict__ wkt, const float* __restrict__ kb,
    const float* __restrict__ vlnw, const float* __restrict__ vlnb,
    const unsigned short* __restrict__ wvt, const float* __restrict__ vb,
    unsigned short* __restrict__ kbuf,      // bf16 [B, 128, 1024] row-major
    unsigned short* __restrict__ vT)        // bf16 [B, 1024, 128] (d-major)
{
    extern __shared__ char smem[];
    unsigned short* xn = (unsigned short*)smem;   // 32 rows x 1536 bf16 (96 KB)
    __shared__ float meanS[32], rstdS[32];
    const int tid = threadIdx.x, w = tid >> 5, lane = tid & 31;
    const int b  = blockIdx.x >> 2;
    const int l0 = (blockIdx.x & 3) * 32;
    const int n = lane & 15, h = lane >> 4;

    // pass 1: LN stats + k-affine normalize (4 rows per wave)
    for (int i = 0; i < 4; ++i) {
        int m = 4 * w + i, l = l0 + m;
        if (l < L_) {
            const float* row = box + ((size_t)b * L_ + l) * DHI_;
            float s = 0.f, sq = 0.f;
            for (int c = lane; c < DHI_; c += 32) { float t = row[c]; s += t; sq += t * t; }
            s = wave_sum(s); sq = wave_sum(sq);
            float mean = s * (1.f / DHI_);
            float var  = sq * (1.f / DHI_) - mean * mean;
            float rstd = rsqrtf(var + 1e-5f);
            if (lane == 0) { meanS[m] = mean; rstdS[m] = rstd; }
            for (int c = lane; c < DHI_; c += 32)
                xn[m * DHI_ + c] = f2bf((row[c] - mean) * rstd * klnw[c] + klnb[c]);
        } else {
            for (int c = lane; c < DHI_; c += 32) xn[m * DHI_ + c] = 0;
        }
    }
    __syncthreads();

    const __bf16* xb = (const __bf16*)xn;
    // k GEMM
    for (int ct = 0; ct < 8; ++ct) {
        int col0 = w * 128 + ct * 16;
        v8f a0 = {}, a1 = {};
        for (int kk = 0; kk < DHI_; kk += 32) {
            v16bf bfr = load_bfrag((const __bf16*)wkt, DHI_, col0, kk, lane);
            v16bf f0  = load_afrag(xb,             DHI_, kk, lane);
            v16bf f1  = load_afrag(xb + 16 * DHI_, DHI_, kk, lane);
            SCHED_FENCE();
            a0 = wmma_bf16(f0, bfr, a0);
            a1 = wmma_bf16(f1, bfr, a1);
        }
        float bv = kb[col0 + n];
        v8f accs[2] = {a0, a1};
        for (int rt = 0; rt < 2; ++rt)
            for (int e = 0; e < 8; ++e) {
                int l = l0 + rt * 16 + e + 8 * h;
                kbuf[((size_t)b * LP_ + l) * DLO_ + col0 + n] = f2bf(accs[rt][e] + bv);
            }
    }
    __syncthreads();

    // pass 2: v-affine normalize (reuse saved stats); padded rows stay zero
    for (int i = 0; i < 4; ++i) {
        int m = 4 * w + i, l = l0 + m;
        if (l < L_) {
            const float* row = box + ((size_t)b * L_ + l) * DHI_;
            float mean = meanS[m], rstd = rstdS[m];
            for (int c = lane; c < DHI_; c += 32)
                xn[m * DHI_ + c] = f2bf((row[c] - mean) * rstd * vlnw[c] + vlnb[c]);
        }
    }
    __syncthreads();

    // v GEMM -> transposed store (vT[d][l]) for K-contiguous B-frags later
    for (int ct = 0; ct < 8; ++ct) {
        int col0 = w * 128 + ct * 16;
        v8f a0 = {}, a1 = {};
        for (int kk = 0; kk < DHI_; kk += 32) {
            v16bf bfr = load_bfrag((const __bf16*)wvt, DHI_, col0, kk, lane);
            v16bf f0  = load_afrag(xb,             DHI_, kk, lane);
            v16bf f1  = load_afrag(xb + 16 * DHI_, DHI_, kk, lane);
            SCHED_FENCE();
            a0 = wmma_bf16(f0, bfr, a0);
            a1 = wmma_bf16(f1, bfr, a1);
        }
        float bv = vb[col0 + n];
        v8f accs[2] = {a0, a1};
        for (int rt = 0; rt < 2; ++rt)
            for (int e = 0; e < 8; ++e) {
                int l = l0 + rt * 16 + e + 8 * h;
                vT[((size_t)b * DLO_ + col0 + n) * LP_ + l] = f2bf(accs[rt][e] + bv);
            }
    }
}

// ---- kernel: attention (q k^T -> masked softmax -> @v, + residual) -------
// grid: B * (N/32) blocks, 256 threads (8 waves).
// dynamic LDS: qt 64 KB | attS 16 KB | attB 8 KB  (88 KB total)
__global__ __launch_bounds__(256) void attn_kernel(
    const unsigned short* __restrict__ qbuf,  // bf16 [B*N, 1024]
    const unsigned short* __restrict__ kbuf,  // bf16 [B, 128, 1024]
    const unsigned short* __restrict__ vT,    // bf16 [B, 1024, 128]
    const int* __restrict__ lengths,          // [B]
    const float* __restrict__ vit,            // [B*N, 1024]
    float* __restrict__ out)                  // [B*N, 1024]
{
    extern __shared__ char smem[];
    unsigned short* qt   = (unsigned short*)smem;                 // 32x1024 bf16
    float*          attS = (float*)(smem + 32 * DLO_ * 2);        // 32x128 f32
    unsigned short* attB = (unsigned short*)(smem + 32 * DLO_ * 2 + 32 * LP_ * 4);
    const int tid = threadIdx.x, w = tid >> 5, lane = tid & 31;
    const int b  = blockIdx.x / (N_ / 32);
    const int nt = blockIdx.x % (N_ / 32);
    const size_t qbase = ((size_t)b * N_ + (size_t)nt * 32) * DLO_;
    const int n = lane & 15, h = lane >> 4;

    // stage the 32x1024 bf16 q tile into LDS
    {
        const f32x4* src = (const f32x4*)(qbuf + qbase);
        f32x4* dst = (f32x4*)qt;
        for (int i = tid; i < 32 * DLO_ * 2 / 16; i += 256) dst[i] = src[i];
    }
    __syncthreads();

    // scores: wave w computes att[:, 16w .. 16w+15] for both row-subtiles
    {
        const __bf16* qb16 = (const __bf16*)qt;
        const __bf16* kb16 = (const __bf16*)(kbuf + (size_t)b * LP_ * DLO_);
        int col0 = w * 16;
        v8f a0 = {}, a1 = {};
        for (int kk = 0; kk < DLO_; kk += 32) {
            v16bf bfr = load_bfrag(kb16, DLO_, col0, kk, lane);   // k rows as B
            v16bf f0  = load_afrag(qb16,             DLO_, kk, lane);
            v16bf f1  = load_afrag(qb16 + 16 * DLO_, DLO_, kk, lane);
            SCHED_FENCE();
            a0 = wmma_bf16(f0, bfr, a0);
            a1 = wmma_bf16(f1, bfr, a1);
        }
        const float scale = 0.03125f;  // 1/sqrt(1024)
        for (int e = 0; e < 8; ++e) {
            attS[(e + 8 * h) * LP_ + col0 + n]      = a0[e] * scale;
            attS[(16 + e + 8 * h) * LP_ + col0 + n] = a1[e] * scale;
        }
    }
    __syncthreads();

    // masked softmax per row (fp32), write bf16 probs; invalid cols -> 0
    if (tid < 32) {
        int len = lengths[b];
        if (len <= 0) {
            // reference: all cols -> finite-min -> uniform over the 100 real cols
            unsigned short u = f2bf(0.01f);
            for (int c = 0; c < LP_; ++c) attB[tid * LP_ + c] = (c < L_) ? u : (unsigned short)0;
        } else {
            if (len > L_) len = L_;
            float mx = -3.0e38f;
            for (int c = 0; c < len; ++c) mx = fmaxf(mx, attS[tid * LP_ + c]);
            float sum = 0.f;
            for (int c = 0; c < len; ++c) {
                float e = __expf(attS[tid * LP_ + c] - mx);
                attS[tid * LP_ + c] = e; sum += e;
            }
            float inv = 1.f / sum;
            for (int c = 0; c < LP_; ++c)
                attB[tid * LP_ + c] = (c < len) ? f2bf(attS[tid * LP_ + c] * inv)
                                                : (unsigned short)0;
        }
    }
    __syncthreads();

    // out = att @ v  (K = 128), fused residual add, fp32 store
    {
        const __bf16* ab16 = (const __bf16*)attB;
        const __bf16* vb16 = (const __bf16*)(vT + (size_t)b * DLO_ * LP_);
        for (int ct = 0; ct < 8; ++ct) {
            int col0 = w * 128 + ct * 16;
            v8f a0 = {}, a1 = {};
            for (int kk = 0; kk < LP_; kk += 32) {
                v16bf bfr = load_bfrag(vb16, LP_, col0, kk, lane);
                v16bf f0  = load_afrag(ab16,            LP_, kk, lane);
                v16bf f1  = load_afrag(ab16 + 16 * LP_, LP_, kk, lane);
                SCHED_FENCE();
                a0 = wmma_bf16(f0, bfr, a0);
                a1 = wmma_bf16(f1, bfr, a1);
            }
            v8f accs[2] = {a0, a1};
            for (int rt = 0; rt < 2; ++rt)
                for (int e = 0; e < 8; ++e) {
                    size_t idx = qbase + (size_t)(rt * 16 + e + 8 * h) * DLO_ + col0 + n;
                    out[idx] = vit[idx] + accs[rt][e];
                }
        }
    }
}

// ---- host launch ---------------------------------------------------------

extern "C" void kernel_launch(void* const* d_in, const int* in_sizes, int n_in,
                              void* d_out, int out_size, void* d_ws, size_t ws_size,
                              hipStream_t stream) {
    const float* vit   = (const float*)d_in[0];
    const float* box   = (const float*)d_in[1];
    const int*   lens  = (const int*)d_in[2];
    const float* q_lnw = (const float*)d_in[3];
    const float* q_lnb = (const float*)d_in[4];
    const float* q_w   = (const float*)d_in[5];
    const float* q_b   = (const float*)d_in[6];
    const float* k_lnw = (const float*)d_in[7];
    const float* k_lnb = (const float*)d_in[8];
    const float* k_w   = (const float*)d_in[9];
    const float* k_b   = (const float*)d_in[10];
    const float* v_lnw = (const float*)d_in[11];
    const float* v_lnb = (const float*)d_in[12];
    const float* v_w   = (const float*)d_in[13];
    const float* v_b   = (const float*)d_in[14];

    char* ws = (char*)d_ws;
    unsigned short* wq_t = (unsigned short*)ws; ws += (size_t)DLO_ * DLO_ * 2;      // 2 MiB
    unsigned short* wk_t = (unsigned short*)ws; ws += (size_t)DHI_ * DLO_ * 2;      // 3 MiB
    unsigned short* wv_t = (unsigned short*)ws; ws += (size_t)DHI_ * DLO_ * 2;      // 3 MiB
    unsigned short* qbuf = (unsigned short*)ws; ws += (size_t)B_ * N_ * DLO_ * 2;   // 36 MiB
    unsigned short* kbuf = (unsigned short*)ws; ws += (size_t)B_ * LP_ * DLO_ * 2;  // 8 MiB
    unsigned short* vTb  = (unsigned short*)ws;                                     // 8 MiB

    // weight prep (transpose + bf16)
    wprep_kernel<<<(DLO_ * 1024) / 256, 256, 0, stream>>>(q_w, wq_t, DLO_);
    wprep_kernel<<<(DHI_ * 1024) / 256, 256, 0, stream>>>(k_w, wk_t, DHI_);
    wprep_kernel<<<(DHI_ * 1024) / 256, 256, 0, stream>>>(v_w, wv_t, DHI_);

    // k & v projections: 128 blocks, 96 KB dynamic LDS
    kv_proj_kernel<<<B_ * 4, 256, 32 * DHI_ * 2, stream>>>(
        box, k_lnw, k_lnb, wk_t, k_b, v_lnw, v_lnb, wv_t, v_b, kbuf, vTb);

    // q projection: 288 blocks, 128 KB dynamic LDS
    q_proj_kernel<<<(B_ * N_) / 64, 256, 64 * DLO_ * 2, stream>>>(
        vit, q_lnw, q_lnb, wq_t, q_b, qbuf);

    // attention + residual: 576 blocks, 88 KB dynamic LDS
    attn_kernel<<<B_ * (N_ / 32), 256, 32 * DLO_ * 2 + 32 * LP_ * 4 + 32 * LP_ * 2, stream>>>(
        qbuf, kbuf, vTb, lens, vit, (float*)d_out);
}
